// RelPosMultiHeadAttention_12360915878708
// MI455X (gfx1250) — matrix operand order repaired and checked by reading, not compiled, not run on hardware
//
#include <hip/hip_runtime.h>
#include <stdint.h>

#define TSEQ 2048
#define HIDN 1024
#define NB   2
#define NH   16
#define HD   64

typedef _Float16 v8h  __attribute__((ext_vector_type(8)));
typedef _Float16 v16h __attribute__((ext_vector_type(16)));
typedef float    v4f  __attribute__((ext_vector_type(4)));
typedef float    v8f  __attribute__((ext_vector_type(8)));

static __device__ __forceinline__ v8f wmma16(v16h a, v16h b, v8f c) {
  // D = A(16x32 f16) * B(32x16 f16) + C(16x16 f32)
  return __builtin_amdgcn_wmma_f32_16x16x32_f16(false, a, false, b, (short)0, c, false, false);
}

// A-fragment (16x32, f16 source, row-major [*, lda]):
// lane<16: K = k0+0..7 (elems 0-7), k0+16..23 (elems 8-15); lane>=16: +8 / +24
static __device__ __forceinline__ v16h load_a16(const _Float16* A, int lda, int row, int k0, int lane) {
  const _Float16* p = A + (size_t)row * lda + k0 + ((lane & 16) ? 8 : 0);
  v8h lo = *(const v8h*)p;
  v8h hi = *(const v8h*)(p + 16);
  v16h a;
#pragma unroll
  for (int i = 0; i < 8; ++i) { a[i] = lo[i]; a[i + 8] = hi[i]; }
  return a;
}

// Same A-fragment but from an f32 source (attention probabilities), cvt to f16.
static __device__ __forceinline__ v16h load_a32(const float* A, int lda, int row, int k0, int lane) {
  const float* p = A + (size_t)row * lda + k0 + ((lane & 16) ? 8 : 0);
  v4f x0 = *(const v4f*)(p);
  v4f x1 = *(const v4f*)(p + 4);
  v4f x2 = *(const v4f*)(p + 16);
  v4f x3 = *(const v4f*)(p + 20);
  v16h a;
#pragma unroll
  for (int i = 0; i < 4; ++i) {
    a[i]      = (_Float16)x0[i];
    a[i + 4]  = (_Float16)x1[i];
    a[i + 8]  = (_Float16)x2[i];
    a[i + 12] = (_Float16)x3[i];
  }
  return a;
}

// B-fragment (32x16): B[k,n] = Bsrc[n,k] with Bsrc row-major [N, ldb].
// lane<16: K = k0+0..7 (elems 0-7), k0+16..23 (elems 8-15); lane>=16: +8 / +24
static __device__ __forceinline__ v16h load_b16(const _Float16* Bm, int ldb, int col, int k0, int lane) {
  const _Float16* p = Bm + (size_t)col * ldb + k0 + ((lane & 16) ? 8 : 0);
  v8h lo = *(const v8h*)p;
  v8h hi = *(const v8h*)(p + 16);
  v16h b;
#pragma unroll
  for (int i = 0; i < 8; ++i) { b[i] = lo[i]; b[i + 8] = hi[i]; }
  return b;
}

__global__ void cvt_f32_f16(const float* __restrict__ in, _Float16* __restrict__ out, int n) {
  int i = blockIdx.x * blockDim.x + threadIdx.x;
  int stride = gridDim.x * blockDim.x;
  for (; i < n; i += stride) out[i] = (_Float16)in[i];
}

// C[M,N] = A[M,K] @ Bsrc[N,K]^T, f16 in / f32 accumulate. One wave = 16x64 strip.
// mode 0: f16 store; mode 1: +bias[col], f16 store;
// mode 2: f16 store transposed per head -> vT[((b*NH+h)*HD+d)*TSEQ + s]; mode 3: f32 store.
__global__ void gemm_nt(const _Float16* __restrict__ A, int lda,
                        const _Float16* __restrict__ Bm, int ldb,
                        int M, int N, int K, int mode,
                        _Float16* __restrict__ Ch, float* __restrict__ Cf,
                        const float* __restrict__ bias, int ldc) {
  int lane = threadIdx.x & 31;
  int wid = blockIdx.x * (blockDim.x >> 5) + (threadIdx.x >> 5);
  int ngrp = N >> 6;
  if (wid >= (M >> 4) * ngrp) return;   // wave-uniform
  int m0 = (wid / ngrp) << 4;
  int n0 = (wid % ngrp) << 6;
  v8f acc[4];
#pragma unroll
  for (int j = 0; j < 4; ++j)
#pragma unroll
    for (int i = 0; i < 8; ++i) acc[j][i] = 0.0f;

  int arow = m0 + (lane & 15);
  for (int k0 = 0; k0 < K; k0 += 32) {
    v16h a = load_a16(A, lda, arow, k0, lane);
#pragma unroll
    for (int j = 0; j < 4; ++j) {
      v16h b = load_b16(Bm, ldb, n0 + 16 * j + (lane & 15), k0, lane);
      acc[j] = wmma16(a, b, acc[j]);
    }
  }
  int nl = lane & 15;
  int mb = m0 + ((lane & 16) ? 8 : 0);
#pragma unroll
  for (int j = 0; j < 4; ++j) {
#pragma unroll
    for (int r = 0; r < 8; ++r) {
      int row = mb + r, col = n0 + 16 * j + nl;
      float v = acc[j][r];
      if (mode == 1) v += bias[col];
      if (mode == 2) {
        int bb = row / TSEQ, s = row % TSEQ, h = col >> 6, d = col & 63;
        Ch[(((size_t)bb * NH + h) * HD + d) * TSEQ + s] = (_Float16)v;
      } else if (mode == 3) {
        Cf[(size_t)row * ldc + col] = v;
      } else {
        Ch[(size_t)row * ldc + col] = (_Float16)v;
      }
    }
  }
}

// S[bh, t, s] = scale * (q_u[b,t,h,:] . k[b,s,h,:])  (writes every (t,s))
__global__ void content_scores(const _Float16* __restrict__ qu,
                               const _Float16* __restrict__ kh,
                               float* __restrict__ S, float scale) {
  int lane = threadIdx.x & 31;
  int wid = blockIdx.x * (blockDim.x >> 5) + (threadIdx.x >> 5);
  int sgrp  = wid & 31;
  int ttile = (wid >> 5) & 127;
  int bh    = wid >> 12;
  if (bh >= NB * NH) return;
  int b = bh >> 4, h = bh & 15;
  int t0 = ttile << 4, s0 = sgrp << 6;
  const _Float16* Aq = qu + (size_t)b * TSEQ * HIDN + h * HD;
  const _Float16* Bk = kh + (size_t)b * TSEQ * HIDN + h * HD;
  v8f acc[4];
#pragma unroll
  for (int j = 0; j < 4; ++j)
#pragma unroll
    for (int i = 0; i < 8; ++i) acc[j][i] = 0.0f;
  int arow = t0 + (lane & 15);
  v16h a0 = load_a16(Aq, HIDN, arow, 0, lane);
  v16h a1 = load_a16(Aq, HIDN, arow, 32, lane);
#pragma unroll
  for (int j = 0; j < 4; ++j) {
    int col = s0 + 16 * j + (lane & 15);
    acc[j] = wmma16(a0, load_b16(Bk, HIDN, col, 0, lane), acc[j]);
    acc[j] = wmma16(a1, load_b16(Bk, HIDN, col, 32, lane), acc[j]);
  }
  float* Sp = S + (size_t)bh * TSEQ * TSEQ;
  int nl = lane & 15, hi = (lane & 16) ? 8 : 0;
#pragma unroll
  for (int j = 0; j < 4; ++j)
#pragma unroll
    for (int r = 0; r < 8; ++r)
      Sp[(size_t)(t0 + hi + r) * TSEQ + (s0 + 16 * j + nl)] = acc[j][r] * scale;
}

// P[t,l] = q_v . rel_k[l], then inverse rel_shift scatter-add into S:
//   l >= T-1-t          : S[t,   l+t-T+1] += v
//   l <  T-1-t && t >= 1: S[t-1, l+t+1 ] += v      (each dest touched at most once)
__global__ void pos_scores(const _Float16* __restrict__ qv,
                           const _Float16* __restrict__ relk,
                           float* __restrict__ S, float scale) {
  int lane = threadIdx.x & 31;
  int wid = blockIdx.x * (blockDim.x >> 5) + (threadIdx.x >> 5);
  int lgrp  = wid & 31;
  int ttile = (wid >> 5) & 127;
  int bh    = wid >> 12;
  if (bh >= NB * NH) return;
  int b = bh >> 4, h = bh & 15;
  int t0 = ttile << 4, l0 = lgrp << 6;
  const _Float16* Aq = qv + (size_t)b * TSEQ * HIDN + h * HD;
  const _Float16* Br = relk + h * HD;
  v8f acc[4];
#pragma unroll
  for (int j = 0; j < 4; ++j)
#pragma unroll
    for (int i = 0; i < 8; ++i) acc[j][i] = 0.0f;
  int arow = t0 + (lane & 15);
  v16h a0 = load_a16(Aq, HIDN, arow, 0, lane);
  v16h a1 = load_a16(Aq, HIDN, arow, 32, lane);
#pragma unroll
  for (int j = 0; j < 4; ++j) {
    int col = l0 + 16 * j + (lane & 15);
    acc[j] = wmma16(a0, load_b16(Br, HIDN, col, 0, lane), acc[j]);
    acc[j] = wmma16(a1, load_b16(Br, HIDN, col, 32, lane), acc[j]);
  }
  float* Sp = S + (size_t)bh * TSEQ * TSEQ;
  int nl = lane & 15, hi = (lane & 16) ? 8 : 0;
#pragma unroll
  for (int j = 0; j < 4; ++j) {
#pragma unroll
    for (int r = 0; r < 8; ++r) {
      int t = t0 + hi + r;
      int l = l0 + 16 * j + nl;
      float v = acc[j][r] * scale;
      if (l >= TSEQ - 1 - t) {
        int s = l + t - (TSEQ - 1);
        Sp[(size_t)t * TSEQ + s] += v;
      } else if (t >= 1) {
        int s = l + t + 1;
        Sp[(size_t)(t - 1) * TSEQ + s] += v;
      }
    }
  }
}

__global__ void softmax_rows(float* __restrict__ S) {
  __shared__ float sm[256];
  float* p = S + (size_t)blockIdx.x * TSEQ;
  int tid = threadIdx.x;
  float v[8];
  float mx = -3.4e38f;
#pragma unroll
  for (int i = 0; i < 8; ++i) { v[i] = p[tid + (i << 8)]; mx = fmaxf(mx, v[i]); }
  sm[tid] = mx; __syncthreads();
  for (int s = 128; s > 0; s >>= 1) {
    if (tid < s) sm[tid] = fmaxf(sm[tid], sm[tid + s]);
    __syncthreads();
  }
  mx = sm[0]; __syncthreads();
  float sum = 0.0f;
#pragma unroll
  for (int i = 0; i < 8; ++i) { v[i] = __expf(v[i] - mx); sum += v[i]; }
  sm[tid] = sum; __syncthreads();
  for (int s = 128; s > 0; s >>= 1) {
    if (tid < s) sm[tid] += sm[tid + s];
    __syncthreads();
  }
  float inv = 1.0f / sm[0];
#pragma unroll
  for (int i = 0; i < 8; ++i) p[tid + (i << 8)] = v[i] * inv;
}

// outh[b*T+t, h*64+d] = sum_s attn[t,s] * v[s,d]   (A from f32 probs, B from vT)
__global__ void attn_v(const float* __restrict__ S, const _Float16* __restrict__ vT,
                       _Float16* __restrict__ outh) {
  int lane = threadIdx.x & 31;
  int wid = blockIdx.x * (blockDim.x >> 5) + (threadIdx.x >> 5);
  int ttile = wid & 127;
  int bh = wid >> 7;
  if (bh >= NB * NH) return;
  int t0 = ttile << 4;
  const float* Sp = S + (size_t)bh * TSEQ * TSEQ;
  const _Float16* Vp = vT + (size_t)bh * HD * TSEQ;
  v8f acc[4];
#pragma unroll
  for (int j = 0; j < 4; ++j)
#pragma unroll
    for (int i = 0; i < 8; ++i) acc[j][i] = 0.0f;
  int arow = t0 + (lane & 15);
  for (int k0 = 0; k0 < TSEQ; k0 += 32) {
    v16h a = load_a32(Sp, TSEQ, arow, k0, lane);
#pragma unroll
    for (int j = 0; j < 4; ++j)
      acc[j] = wmma16(a, load_b16(Vp, TSEQ, 16 * j + (lane & 15), k0, lane), acc[j]);
  }
  int b = bh >> 4, h = bh & 15;
  _Float16* Op = outh + (size_t)b * TSEQ * HIDN + h * HD;
  int nl = lane & 15, hi = (lane & 16) ? 8 : 0;
#pragma unroll
  for (int j = 0; j < 4; ++j)
#pragma unroll
    for (int r = 0; r < 8; ++r)
      Op[(size_t)(t0 + hi + r) * HIDN + 16 * j + nl] = (_Float16)acc[j][r];
}

extern "C" void kernel_launch(void* const* d_in, const int* in_sizes, int n_in,
                              void* d_out, int out_size, void* d_ws, size_t ws_size,
                              hipStream_t stream) {
  const float* x      = (const float*)d_in[0];
  const float* pos    = (const float*)d_in[1];
  const float* Wq     = (const float*)d_in[2];
  const float* Wk     = (const float*)d_in[3];
  const float* Wv     = (const float*)d_in[4];
  const float* Wo     = (const float*)d_in[5];
  const float* Wrel   = (const float*)d_in[6];
  const float* bias_u = (const float*)d_in[7];
  const float* bias_v = (const float*)d_in[8];
  float* out = (float*)d_out;

  char* ws = (char*)d_ws;
  size_t off = 0;
  auto carve = [&](size_t bytes) -> void* {
    void* p = ws + off;
    off = (off + bytes + 255) & ~(size_t)255;
    return p;
  };
  const size_t nx = (size_t)NB * TSEQ * HIDN;      // 4M
  const size_t nw = (size_t)HIDN * HIDN;           // 1M
  const size_t np = (size_t)TSEQ * HIDN;           // 2M
  _Float16* xh    = (_Float16*)carve(nx * 2);
  _Float16* Wqh   = (_Float16*)carve(nw * 2);
  _Float16* Wkh   = (_Float16*)carve(nw * 2);
  _Float16* Wvh   = (_Float16*)carve(nw * 2);
  _Float16* Wrelh = (_Float16*)carve(nw * 2);
  _Float16* Woh   = (_Float16*)carve(nw * 2);
  _Float16* poseh = (_Float16*)carve(np * 2);
  _Float16* qu    = (_Float16*)carve(nx * 2);
  _Float16* qv    = (_Float16*)carve(nx * 2);
  _Float16* kh    = (_Float16*)carve(nx * 2);
  _Float16* vT    = (_Float16*)carve(nx * 2);
  _Float16* relk  = (_Float16*)carve(np * 2);
  _Float16* outh  = (_Float16*)carve(nx * 2);
  float*    S     = (float*)carve((size_t)NB * NH * TSEQ * TSEQ * 4);  // 512 MB

  // 1) f32 -> f16 conversions
  cvt_f32_f16<<<2048, 256, 0, stream>>>(x,    xh,    (int)nx);
  cvt_f32_f16<<<1024, 256, 0, stream>>>(pos,  poseh, (int)np);
  cvt_f32_f16<<<512,  256, 0, stream>>>(Wq,   Wqh,   (int)nw);
  cvt_f32_f16<<<512,  256, 0, stream>>>(Wk,   Wkh,   (int)nw);
  cvt_f32_f16<<<512,  256, 0, stream>>>(Wv,   Wvh,   (int)nw);
  cvt_f32_f16<<<512,  256, 0, stream>>>(Wrel, Wrelh, (int)nw);
  cvt_f32_f16<<<512,  256, 0, stream>>>(Wo,   Woh,   (int)nw);

  // 2) projections (NT GEMMs):  waves = (M/16)*(N/64), 4 waves/block
  const int M1 = NB * TSEQ;
  const int blkP = (M1 / 16) * (HIDN / 64) / 4;   // 1024
  gemm_nt<<<blkP, 128, 0, stream>>>(xh, HIDN, Wqh, HIDN, M1, HIDN, HIDN, 1, qu, nullptr, bias_u, HIDN);
  gemm_nt<<<blkP, 128, 0, stream>>>(xh, HIDN, Wqh, HIDN, M1, HIDN, HIDN, 1, qv, nullptr, bias_v, HIDN);
  gemm_nt<<<blkP, 128, 0, stream>>>(xh, HIDN, Wkh, HIDN, M1, HIDN, HIDN, 0, kh, nullptr, nullptr, HIDN);
  gemm_nt<<<blkP, 128, 0, stream>>>(xh, HIDN, Wvh, HIDN, M1, HIDN, HIDN, 2, vT, nullptr, nullptr, HIDN);
  const int blkR = (TSEQ / 16) * (HIDN / 64) / 4; // 512
  gemm_nt<<<blkR, 128, 0, stream>>>(poseh, HIDN, Wrelh, HIDN, TSEQ, HIDN, HIDN, 0, relk, nullptr, nullptr, HIDN);

  // 3) attention
  const float scale = 0.125f;  // 1/sqrt(64)
  const int blkS = NB * NH * (TSEQ / 16) * (TSEQ / 64) / 4;  // 32768
  content_scores<<<blkS, 128, 0, stream>>>(qu, kh, S, scale);
  pos_scores   <<<blkS, 128, 0, stream>>>(qv, relk, S, scale);
  softmax_rows <<<NB * NH * TSEQ, 256, 0, stream>>>(S);
  const int blkAV = NB * NH * (TSEQ / 16) / 4;               // 1024
  attn_v<<<blkAV, 128, 0, stream>>>(S, vT, outh);

  // 4) output projection -> f32 d_out
  gemm_nt<<<blkP, 128, 0, stream>>>(outh, HIDN, Woh, HIDN, M1, HIDN, HIDN, 3, nullptr, out, nullptr, HIDN);

  (void)in_sizes; (void)n_in; (void)out_size; (void)ws_size;
}
